// KANLayer_21638045237976
// MI455X (gfx1250) — matrix operand order repaired and checked by reading, not compiled, not run on hardware
//
#include <hip/hip_runtime.h>
#include <hip/hip_bf16.h>

typedef __attribute__((ext_vector_type(16))) __bf16 v16bf;
typedef __attribute__((ext_vector_type(8)))  float  v8f;

#define KAN_GRID   16
#define IN_DIM     256
#define OUT_DIM    128
#define BATCH      4096
#define K_SPLINE   (IN_DIM * KAN_GRID)    // 4096 hat-basis features
#define K_TOTAL    (K_SPLINE + IN_DIM)    // + 256 silu base features = 4352
#define M_TILE     128
#define N_TILE     64
#define KB         64                     // k per LDS stage (2 WMMA k-slices)
#define NSTAGE     (K_TOTAL / KB)         // 68 = 64 spline + 4 base stages
#define A_STRIDE   72                     // ushorts/row (144B: 16B-aligned, conflict-free)

static __device__ __forceinline__ unsigned short f2bf(float f) {
    __bf16 h = (__bf16)f;
    return __builtin_bit_cast(unsigned short, h);
}

// ---------------------------------------------------------------------------
// Kernel 1: fold spline_scale/base_scale into a k-major bf16 weight matrix
//   Wt[o][k], k in [0,4096): spline_scale[o,i]*sv[o,i,g]  (k = i*16+g)
//             k in [4096,4352): base_scale[o, k-4096]
// ---------------------------------------------------------------------------
__global__ __launch_bounds__(256) void kan_build_w(
    const float* __restrict__ spline_values,
    const float* __restrict__ base_scale,
    const float* __restrict__ spline_scale,
    unsigned short* __restrict__ Wt)
{
    const int o = blockIdx.x;
    unsigned short* wrow = Wt + (size_t)o * K_TOTAL;
    for (int k = threadIdx.x; k < K_SPLINE; k += 256) {
        const int i = k >> 4;
        wrow[k] = f2bf(spline_scale[o * IN_DIM + i] * spline_values[o * K_SPLINE + k]);
    }
    for (int i = threadIdx.x; i < IN_DIM; i += 256) {
        wrow[K_SPLINE + i] = f2bf(base_scale[o * IN_DIM + i]);
    }
}

// ---------------------------------------------------------------------------
// Build one 128x64 bf16 activation tile (stage) into LDS.
//   Spline stage: 4 input dims, 16 hat-basis weights each (2 nonzeros).
//   Base stage:   64 silu(x) features.
// 512 (row, quarter) tasks over 256 threads.
// ---------------------------------------------------------------------------
static __device__ __forceinline__ void build_stage(
    int stage, unsigned short* __restrict__ buf,
    const float* __restrict__ x, int m0, int tid)
{
    const int k0 = stage * KB;
#pragma unroll
    for (int rpt = 0; rpt < 2; ++rpt) {
        const int task = tid + rpt * 256;       // 0..511
        const int row  = task >> 2;             // 0..127
        const int q    = task & 3;              // 16-feature quarter
        unsigned short* dst = &buf[row * A_STRIDE + q * 16];
        if (k0 < K_SPLINE) {
            const int i  = (k0 >> 4) + q;
            const float xv = x[(m0 + row) * IN_DIM + i];
            const float xc = fminf(fmaxf(xv, -2.0f), 2.0f);
            const float t  = (xc + 2.0f) * (15.0f * 0.25f);
            int left = (int)floorf(t);
            left = left > (KAN_GRID - 2) ? (KAN_GRID - 2) : left;
            const float alpha = t - (float)left;
            const uint4 z = make_uint4(0u, 0u, 0u, 0u);
            ((uint4*)dst)[0] = z;
            ((uint4*)dst)[1] = z;
            dst[left]     = f2bf(1.0f - alpha);
            dst[left + 1] = f2bf(alpha);
        } else {
            const int c0 = (k0 - K_SPLINE) + q * 16;
            const float4* xs = (const float4*)&x[(m0 + row) * IN_DIM + c0];
            unsigned int p[8];
#pragma unroll
            for (int qq = 0; qq < 4; ++qq) {
                const float4 v = xs[qq];
                const float s0 = v.x / (1.0f + __expf(-v.x));
                const float s1 = v.y / (1.0f + __expf(-v.y));
                const float s2 = v.z / (1.0f + __expf(-v.z));
                const float s3 = v.w / (1.0f + __expf(-v.w));
                p[2*qq]   = (unsigned)f2bf(s0) | ((unsigned)f2bf(s1) << 16);
                p[2*qq+1] = (unsigned)f2bf(s2) | ((unsigned)f2bf(s3) << 16);
            }
            ((uint4*)dst)[0] = make_uint4(p[0], p[1], p[2], p[3]);
            ((uint4*)dst)[1] = make_uint4(p[4], p[5], p[6], p[7]);
        }
    }
}

// ---------------------------------------------------------------------------
// Kernel 2: out = f(x) @ Wt^T via bf16 WMMA, double-buffered LDS A-tiles.
//   WG = 256 threads (8 wave32), 128x64 output tile, wave = 32x32 patch.
//   One barrier per 64-k stage; build of stage s+1 overlaps WMMAs of stage s.
// ---------------------------------------------------------------------------
__global__ __launch_bounds__(256) void kan_gemm(
    const float* __restrict__ x,
    const unsigned short* __restrict__ Wt,
    float* __restrict__ out)
{
    __shared__ unsigned short aT[2][M_TILE * A_STRIDE];  // 2 x 18 KB ping-pong

    const int tid  = threadIdx.x;
    const int lane = tid & 31;
    const int wave = tid >> 5;
    const int wm   = wave & 3;        // 4 x 32-row bands
    const int wn   = wave >> 2;       // 2 x 32-col bands
    const int l15  = lane & 15;
    const int lHi  = lane >> 4;       // lane half selects the K-subslice
    const int m0   = blockIdx.x * M_TILE;
    const int n0   = blockIdx.y * N_TILE;

    v8f acc[2][2];
    acc[0][0] = v8f{}; acc[0][1] = v8f{}; acc[1][0] = v8f{}; acc[1][1] = v8f{};

    union Frag { uint4 u[2]; v16bf v; };

    build_stage(0, aT[0], x, m0, tid);
    __syncthreads();

    for (int s = 0; s < NSTAGE; ++s) {
        const int cur = s & 1;
        const int k0  = s * KB;

        // ---- A fragments from current LDS buffer (2 k-slices x 2 M-tiles) --
        Frag afr[2][2];
#pragma unroll
        for (int sm = 0; sm < 2; ++sm) {
            const int row = wm * 32 + sm * 16 + l15;
#pragma unroll
            for (int kk = 0; kk < 2; ++kk) {
                const uint4* ap =
                    (const uint4*)&aT[cur][row * A_STRIDE + kk * 32 + lHi * 8];
                afr[sm][kk].u[0] = ap[0];   // K 0..7 / 8..15 of slice
                afr[sm][kk].u[1] = ap[2];   // K 16..23 / 24..31 of slice
            }
        }

        // ---- B fragments: two 16-byte chunks per lane (K 8h..8h+7, 16+8h..) from k-major Wt ----
        Frag bfr[2][2];
#pragma unroll
        for (int sn = 0; sn < 2; ++sn) {
            const int col = n0 + wn * 32 + sn * 16 + l15;   // B column = lane
#pragma unroll
            for (int kk = 0; kk < 2; ++kk) {
                const uint4* bp = (const uint4*)
                    (Wt + (size_t)col * K_TOTAL + k0 + kk * 32 + lHi * 8);
                bfr[sn][kk].u[0] = bp[0];
                bfr[sn][kk].u[1] = bp[2];
            }
        }

        // ---- build next stage into the other buffer (overlaps WMMAs) -----
        if (s + 1 < NSTAGE)
            build_stage(s + 1, aT[cur ^ 1], x, m0, tid);

        // prefetch the following W k-slice toward L2/L0
        if (s + 1 < NSTAGE) {
            const unsigned short* pf =
                Wt + (size_t)(n0 + wn * 32 + l15) * K_TOTAL + (k0 + KB);
            __builtin_prefetch(pf, 0, 1);
        }

        // ---- 8 WMMAs per wave per stage: v_wmma_f32_16x16x32_bf16 --------
#pragma unroll
        for (int kk = 0; kk < 2; ++kk)
#pragma unroll
            for (int sm = 0; sm < 2; ++sm)
#pragma unroll
                for (int sn = 0; sn < 2; ++sn)
                    acc[sm][sn] = __builtin_amdgcn_wmma_f32_16x16x32_bf16(
                        false, afr[sm][kk].v, false, bfr[sn][kk].v,
                        (short)0, acc[sm][sn], false, false);

        __syncthreads();   // next buffer fully built; current reads retired
    }

    // ---- store D per ISA C/D layout: VGPR r -> rows r (lanes 0-15), r+8 ----
#pragma unroll
    for (int sm = 0; sm < 2; ++sm) {
#pragma unroll
        for (int sn = 0; sn < 2; ++sn) {
            const int col = n0 + wn * 32 + sn * 16 + l15;
#pragma unroll
            for (int r = 0; r < 8; ++r) {
                const int row = m0 + wm * 32 + sm * 16 + r + lHi * 8;
                out[row * OUT_DIM + col] = acc[sm][sn][r];
            }
        }
    }
}

extern "C" void kernel_launch(void* const* d_in, const int* in_sizes, int n_in,
                              void* d_out, int out_size, void* d_ws, size_t ws_size,
                              hipStream_t stream) {
    (void)in_sizes; (void)n_in; (void)out_size; (void)ws_size;
    const float* x             = (const float*)d_in[0];
    const float* spline_values = (const float*)d_in[1];
    const float* base_scale    = (const float*)d_in[2];
    const float* spline_scale  = (const float*)d_in[3];
    float* out = (float*)d_out;
    unsigned short* Wt = (unsigned short*)d_ws;   // needs 4352*128*2 B ~= 1.06 MB

    kan_build_w<<<dim3(OUT_DIM), dim3(256), 0, stream>>>(
        spline_values, base_scale, spline_scale, Wt);
    kan_gemm<<<dim3(BATCH / M_TILE, OUT_DIM / N_TILE), dim3(256), 0, stream>>>(
        x, Wt, out);
}
